// FixupResNet_49830210568338
// MI455X (gfx1250) — matrix operand order repaired and checked by reading, not compiled, not run on hardware
//
#include <hip/hip_runtime.h>

typedef __bf16 bf16_t;
typedef __attribute__((ext_vector_type(16))) __bf16 v16bf;
typedef __attribute__((ext_vector_type(8)))  float v8f;
typedef __attribute__((__vector_size__(4 * sizeof(int)))) int v4i;

#define THREADS 256

#if __has_builtin(__builtin_amdgcn_global_load_async_to_lds_b128)
#define HAVE_ASYNC 1
#else
#define HAVE_ASYNC 0
#endif

// builtin wants generic int4* for both the global source and the LDS dest
#define GPTR(x) ((v4i*)(void*)const_cast<bf16_t*>(x))
#define LPTR(x) ((v4i*)(void*)(x))

__device__ __forceinline__ void async_wait0()
{
#if HAVE_ASYNC
#if __has_builtin(__builtin_amdgcn_s_wait_asynccnt)
  __builtin_amdgcn_s_wait_asynccnt(0);
#else
  asm volatile("s_wait_asynccnt 0" ::: "memory");
#endif
#endif
}

// ---------------------------------------------------------------------------
// Implicit-GEMM convolution, NHWC activations (bf16) x KRSC weights (bf16).
// GEMM view: A = im2col(act) [M = N*Ho*Wo, Ktot = C*R*S]
//            B = weights^T   [Ktot, Kout]
//            D = out          [M, Kout]  (NHWC, channels fastest)
// Block tile: 128 pixels x 64 channels, K-step 32, double-buffered LDS with
// async global->LDS staging overlapped with v_wmma_f32_16x16x32_bf16 math.
// 8 wave32 waves: 4 along M x 2 along N, each wave owns 2x2 16x16 WMMA tiles.
// ---------------------------------------------------------------------------
struct ConvP {
  const bf16_t* __restrict__ in;    // [N,H,W,C] bf16
  const bf16_t* __restrict__ wt;    // [Kout, R,S,C] bf16 (k = (r*S+s)*C+c contiguous)
  bf16_t* __restrict__ outb;        // [M, Kout] bf16 (epi 0/2/3)
  float*  __restrict__ outf;        // [M, Kout] f32  (epi 4)
  const bf16_t* __restrict__ idn;   // residual identity (epi 2)
  const float* preb;                // scalar pre-bias (after relu on A load)
  const float* postb;               // scalar post-bias
  const float* rscale;              // Fixup residual scale (scalar)
  const float* rbias;               // Fixup residual bias (scalar)
  const float* vbias;               // per-channel bias (epi 4, fc_b)
  int N, H, W, C, K, R, S, st, pad, Ho, Wo, prerelu, epi;
};

__global__ __launch_bounds__(THREADS)
void conv_wmma_kernel(ConvP p)
{
  __shared__ bf16_t As[2][128][40];   // [buf][pixel][k] (+8 pad: bank-conflict dodge)
  __shared__ bf16_t Bs[2][64][40];    // [buf][channel][k]

  const int tid  = threadIdx.x;
  const int lane = tid & 31;
  const int wave = tid >> 5;
  const int wm   = wave & 3;          // 32-row strip of the 128-row tile
  const int wn   = wave >> 2;         // 32-col strip of the 64-col tile

  const int M    = p.N * p.Ho * p.Wo;
  const int Ktot = p.C * p.R * p.S;
  const int pix0 = blockIdx.x * 128;
  const int ch0  = blockIdx.y * 64;

  const float preb  = p.preb  ? p.preb[0]  : 0.0f;
  const float postb = p.postb ? p.postb[0] : 0.0f;

  v8f acc[2][2];
#pragma unroll
  for (int i = 0; i < 2; ++i)
#pragma unroll
    for (int j = 0; j < 2; ++j)
#pragma unroll
      for (int e = 0; e < 8; ++e) acc[i][j][e] = 0.0f;

  // A loader role: 2 threads per pixel row, 16 k-elements each
  const int arow = tid & 127;
  const int akk  = (tid >> 7) << 4;
  const int pix  = pix0 + arow;
  int an = 0, aho = 0, awo = 0;
  if (pix < M) { awo = pix % p.Wo; int t = pix / p.Wo; aho = t % p.Ho; an = t / p.Ho; }
  const int hbase = aho * p.st - p.pad;
  const int wbase = awo * p.st - p.pad;
  const bool is11  = (p.R == 1) && (p.S == 1);
  // async fast path: 1x1, no fused pre-op, 16-aligned channel count
  const bool fastA = HAVE_ASYNC && is11 && (p.prerelu == 0) && ((p.C & 15) == 0);
  long abase = 0;
  if (is11 && pix < M)
    abase = (((long)an * p.H + hbase) * p.W + wbase) * p.C;   // pad==0 for 1x1

  // B loader role: 4 threads per output channel, 8 contiguous k each
  const int brow = tid & 63;
  const int bkk  = (tid >> 6) << 3;
  const int ko   = ch0 + brow;
  const bool fastB = HAVE_ASYNC && ((Ktot & 7) == 0);

  const int ksteps = (Ktot + 31) >> 5;

  // ---- tile staging (async b128 -> LDS when possible, scalar gather otherwise)
  auto stageA = [&](int ks, int buf) {
    const int k0 = ks << 5;
#if HAVE_ASYNC
    if (fastA && pix < M) {
      const bf16_t* g = p.in + abase + k0 + akk;
      bf16_t* l = &As[buf][arow][akk];
      __builtin_amdgcn_global_load_async_to_lds_b128(GPTR(g), LPTR(l), 0, 0);
      __builtin_amdgcn_global_load_async_to_lds_b128(GPTR(g + 8), LPTR(l + 8), 0, 0);
      return;
    }
#endif
#pragma unroll 4
    for (int j = 0; j < 16; ++j) {
      const int k = k0 + akk + j;
      float v = 0.0f;
      if (pix < M && k < Ktot) {
        int c, hi, wi;
        if (is11) { c = k; hi = hbase; wi = wbase; }
        else { c = k % p.C; const int rs = k / p.C; hi = hbase + rs / p.S; wi = wbase + rs % p.S; }
        if ((unsigned)hi < (unsigned)p.H && (unsigned)wi < (unsigned)p.W) {
          v = (float)p.in[(((long)an * p.H + hi) * p.W + wi) * p.C + c];
          if (p.prerelu) v = fmaxf(v, 0.0f) + preb;   // zero-pad applies AFTER the pre-op
        }
      }
      As[buf][arow][akk + j] = (bf16_t)v;
    }
  };

  auto stageB = [&](int ks, int buf) {
    const int k0 = ks << 5;
#if HAVE_ASYNC
    if (fastB && ko < p.K && (k0 + bkk + 8) <= Ktot) {
      __builtin_amdgcn_global_load_async_to_lds_b128(
          GPTR(p.wt + (long)ko * Ktot + k0 + bkk), LPTR(&Bs[buf][brow][bkk]), 0, 0);
      return;
    }
#endif
    if (ko < p.K && (k0 + 32 + bkk) < Ktot)
      __builtin_prefetch(&p.wt[(long)ko * Ktot + k0 + 32 + bkk], 0, 0);
#pragma unroll
    for (int j = 0; j < 8; ++j) {
      const int k = k0 + bkk + j;
      float v = 0.0f;
      if (ko < p.K && k < Ktot) v = (float)p.wt[(long)ko * Ktot + k];
      Bs[buf][brow][bkk + j] = (bf16_t)v;
    }
  };

  // ---- software pipeline: prime buffer 0, then overlap stage(n+1) with wmma(n)
  stageA(0, 0);
  stageB(0, 0);
  async_wait0();
  __syncthreads();

  for (int ks = 0; ks < ksteps; ++ks) {
    const int cb = ks & 1;
    if (ks + 1 < ksteps) { stageA(ks + 1, cb ^ 1); stageB(ks + 1, cb ^ 1); }

    // ---- 2x2 WMMA tiles per wave from LDS buffer cb
#pragma unroll
    for (int tm = 0; tm < 2; ++tm) {
      const int row  = wm * 32 + tm * 16 + (lane & 15);   // A: M = lane%16
      const int halo = (lane >> 4) << 4;                  // A: lanes 16-31 -> K+16
      v16bf a;
#pragma unroll
      for (int v2 = 0; v2 < 8; ++v2) {
        // A k order used here (contiguous, matches B): K = half*16 + 2v + j
        const int kk = halo + (v2 << 1);
        a[2 * v2]     = As[cb][row][kk];
        a[2 * v2 + 1] = As[cb][row][kk + 1];
      }
#pragma unroll
      for (int tn = 0; tn < 2; ++tn) {
        const int col = wn * 32 + tn * 16 + (lane & 15);  // B: N = lane%16
        const int kh  = (lane >> 4) << 4;                 // B: lanes 16-31 -> K+16
        v16bf b;
#pragma unroll
        for (int v2 = 0; v2 < 8; ++v2) {
          // B k order used here (contiguous, matches A): K = half*16 + 2v + j
          b[2 * v2]     = Bs[cb][col][kh + 2 * v2];
          b[2 * v2 + 1] = Bs[cb][col][kh + 2 * v2 + 1];
        }
        acc[tm][tn] = __builtin_amdgcn_wmma_f32_16x16x32_bf16(
            false, a, false, b, (short)0, acc[tm][tn], false, false);
      }
    }

    async_wait0();     // own async copies into buf cb^1 landed
    __syncthreads();   // everyone done reading cb / writing cb^1
  }

  // ---- epilogue (C/D layout: VGPR e -> M = half*8 + e, N = lane%16)
  const float rsc = p.rscale ? p.rscale[0] : 1.0f;
  const float rbi = p.rbias  ? p.rbias[0]  : 0.0f;
#pragma unroll
  for (int tm = 0; tm < 2; ++tm) {
#pragma unroll
    for (int tn = 0; tn < 2; ++tn) {
      const int chan = ch0 + wn * 32 + tn * 16 + (lane & 15);
#pragma unroll
      for (int e = 0; e < 8; ++e) {
        const int prow = pix0 + wm * 32 + tm * 16 + ((lane >> 4) << 3) + e;
        if (prow < M && chan < p.K) {
          float v = acc[tm][tn][e] + postb;
          const long oidx = (long)prow * p.K + chan;
          switch (p.epi) {
            case 2: {  // Fixup residual: relu((conv+b)*rscale + rbias + identity)
              const float idv = (float)p.idn[oidx];
              p.outb[oidx] = (bf16_t)fmaxf(v * rsc + rbi + idv, 0.0f);
            } break;
            case 3:    // stem: relu(conv + bias1)
              p.outb[oidx] = (bf16_t)fmaxf(v, 0.0f);
              break;
            case 4:    // FC head: f32 out + per-channel bias
              p.outf[oidx] = v + (p.vbias ? p.vbias[chan] : 0.0f);
              break;
            default:   // plain + scalar bias
              p.outb[oidx] = (bf16_t)v;
          }
        }
      }
    }
  }
}

// ---------------------------------------------------------------------------
// Support kernels
// ---------------------------------------------------------------------------
__global__ void wcvt_kernel(const float* __restrict__ src, bf16_t* __restrict__ dst,
                            int O, int C, int R, int S)
{
  const long tot = (long)O * C * R * S;
  long idx = (long)blockIdx.x * blockDim.x + threadIdx.x;
  if (idx >= tot) return;
  const int c = (int)(idx % C); long t = idx / C;
  const int s = (int)(t % S);   t /= S;
  const int r = (int)(t % R);
  const int o = (int)(t / R);
  // OIHW -> O,R,S,C (GEMM k-index contiguous in memory)
  dst[idx] = (bf16_t)src[(((long)o * C + c) * R + r) * S + s];
}

__global__ void xcvt_kernel(const float* __restrict__ src, bf16_t* __restrict__ dst,
                            int N, int C, int H, int W)
{
  const long tot = (long)N * H * W * C;
  long idx = (long)blockIdx.x * blockDim.x + threadIdx.x;
  if (idx >= tot) return;
  const int c = (int)(idx % C); long t = idx / C;
  const int w = (int)(t % W);   t /= W;
  const int h = (int)(t % H);
  const int n = (int)(t / H);
  dst[idx] = (bf16_t)src[(((long)n * C + c) * H + h) * W + w];   // NCHW -> NHWC
}

__global__ void maxpool_kernel(const bf16_t* __restrict__ in, bf16_t* __restrict__ out,
                               int N, int H, int W, int C, int Ho, int Wo)
{
  const long tot = (long)N * Ho * Wo * C;
  long idx = (long)blockIdx.x * blockDim.x + threadIdx.x;
  if (idx >= tot) return;
  const int c  = (int)(idx % C); long t = idx / C;
  const int wo = (int)(t % Wo);  t /= Wo;
  const int ho = (int)(t % Ho);
  const int n  = (int)(t / Ho);
  float m = -3.0e38f;
  for (int dy = 0; dy < 3; ++dy) {
    const int hi = ho * 2 - 1 + dy;
    if ((unsigned)hi >= (unsigned)H) continue;
    for (int dx = 0; dx < 3; ++dx) {
      const int wi = wo * 2 - 1 + dx;
      if ((unsigned)wi >= (unsigned)W) continue;
      m = fmaxf(m, (float)in[(((long)n * H + hi) * W + wi) * C + c]);
    }
  }
  out[idx] = (bf16_t)m;
}

__global__ void gap_kernel(const bf16_t* __restrict__ in, bf16_t* __restrict__ out,
                           const float* bias2, int N, int C, int HW)
{
  long idx = (long)blockIdx.x * blockDim.x + threadIdx.x;
  if (idx >= (long)N * C) return;
  const int c = (int)(idx % C);
  const int n = (int)(idx / C);
  float s = 0.0f;
  for (int i = 0; i < HW; ++i)
    s += (float)in[((long)n * HW + i) * C + c];
  out[idx] = (bf16_t)(s / (float)HW + bias2[0]);
}

// ---------------------------------------------------------------------------
// Host orchestration
// ---------------------------------------------------------------------------
extern "C" void kernel_launch(void* const* d_in, const int* in_sizes, int n_in,
                              void* d_out, int out_size, void* d_ws, size_t ws_size,
                              hipStream_t stream)
{
  (void)in_sizes; (void)n_in; (void)out_size; (void)ws_size;
  static const int NL[4] = {3, 4, 6, 3};
  static const int PL[4] = {64, 128, 256, 512};
  static const int ST[4] = {1, 2, 2, 2};
  const int B = 32;

  // ---- workspace carving
  char* wsb = (char*)d_ws;
  size_t off = 0;
  auto carve = [&](size_t bytes) -> char* {
    off = (off + 255) & ~(size_t)255;
    char* ptr = wsb + off;
    off += bytes;
    return ptr;
  };
  bf16_t* warena = (bf16_t*)carve((size_t)64 << 20);   // bf16 weights (~51 MB used)
  size_t  woff   = 0;
  bf16_t* xbf    = (bf16_t*)carve((size_t)B * 224 * 224 * 3 * sizeof(bf16_t));
  const size_t PB = (size_t)B * 112 * 112 * 64 * sizeof(bf16_t);  // max activation
  bf16_t* cur     = (bf16_t*)carve(PB);
  bf16_t* pool[3] = { (bf16_t*)carve(PB), (bf16_t*)carve(PB), (bf16_t*)carve(PB) };
  bf16_t* gapb    = (bf16_t*)carve((size_t)B * 2048 * sizeof(bf16_t));

  auto wconv = [&](const float* src, int O, int C, int R, int S) -> bf16_t* {
    bf16_t* dst = warena + woff;
    const long tot = (long)O * C * R * S;
    woff += (size_t)tot;
    wcvt_kernel<<<dim3((unsigned)((tot + 255) / 256)), dim3(256), 0, stream>>>(src, dst, O, C, R, S);
    return dst;
  };

  auto conv = [&](const bf16_t* in, const bf16_t* wt, bf16_t* outb, float* outf,
                  const bf16_t* idn, const float* preb, const float* postb,
                  const float* rscale, const float* rbias, const float* vbias,
                  int N, int H, int W, int C, int K, int R, int S, int st, int pad,
                  int prerelu, int epi) {
    ConvP p;
    p.in = in; p.wt = wt; p.outb = outb; p.outf = outf; p.idn = idn;
    p.preb = preb; p.postb = postb; p.rscale = rscale; p.rbias = rbias; p.vbias = vbias;
    p.N = N; p.H = H; p.W = W; p.C = C; p.K = K; p.R = R; p.S = S; p.st = st; p.pad = pad;
    p.Ho = (H + 2 * pad - R) / st + 1;
    p.Wo = (W + 2 * pad - S) / st + 1;
    p.prerelu = prerelu; p.epi = epi;
    const int M = N * p.Ho * p.Wo;
    dim3 grid((unsigned)((M + 127) / 128), (unsigned)((K + 63) / 64));
    conv_wmma_kernel<<<grid, dim3(THREADS), 0, stream>>>(p);
  };

  // ---- inputs (setup_inputs dict order)
  int ii = 0;
  const float* X      = (const float*)d_in[ii++];
  const float* conv1w = (const float*)d_in[ii++];
  const float* bias1  = (const float*)d_in[ii++];
  const float* bias2  = (const float*)d_in[ii++];

  // ---- input NCHW f32 -> NHWC bf16
  {
    const long tot = (long)B * 224 * 224 * 3;
    xcvt_kernel<<<dim3((unsigned)((tot + 255) / 256)), dim3(256), 0, stream>>>(X, xbf, B, 3, 224, 224);
  }
  // ---- stem: 7x7/2 conv (+bias1, relu fused), then 3x3/2 maxpool
  bf16_t* w0 = wconv(conv1w, 64, 3, 7, 7);
  conv(xbf, w0, pool[0], nullptr, nullptr, nullptr, bias1, nullptr, nullptr, nullptr,
       B, 224, 224, 3, 64, 7, 7, 2, 3, /*prerelu=*/0, /*epi=*/3);
  {
    const long tot = (long)B * 56 * 56 * 64;
    maxpool_kernel<<<dim3((unsigned)((tot + 255) / 256)), dim3(256), 0, stream>>>(
        pool[0], cur, B, 112, 112, 64, 56, 56);
  }

  // ---- Fixup bottleneck stages
  int H = 56, inpl = 64;
  for (int li = 0; li < 4; ++li) {
    for (int bi = 0; bi < NL[li]; ++bi) {
      const int pl = PL[li];
      const int stride = (bi == 0) ? ST[li] : 1;

      const float* w1  = (const float*)d_in[ii++];
      const float* w2  = (const float*)d_in[ii++];
      const float* w3  = (const float*)d_in[ii++];
      const float* b1b = (const float*)d_in[ii++];
      const float* b2a = (const float*)d_in[ii++];
      const float* b2b = (const float*)d_in[ii++];
      const float* b3a = (const float*)d_in[ii++];
      const float* b3b = (const float*)d_in[ii++];
      const float* rbias  = (const float*)d_in[ii++];
      const float* rscale = (const float*)d_in[ii++];
      const float* wd = nullptr;
      if (bi == 0 && (stride != 1 || inpl != pl * 4))
        wd = (const float*)d_in[ii++];

      bf16_t* w1b = wconv(w1, pl, inpl, 1, 1);
      bf16_t* w2b = wconv(w2, pl, pl, 3, 3);
      bf16_t* w3b = wconv(w3, pl * 4, pl, 1, 1);
      bf16_t* wdb = wd ? wconv(wd, pl * 4, inpl, 1, 1) : nullptr;

      const int Ho = (H - 1) / stride + 1;
      bf16_t* t1 = pool[0];
      bf16_t* t2 = pool[1];

      // t1 = conv1x1(x) + b1b                      (async A + async B path)
      conv(cur, w1b, t1, nullptr, nullptr, nullptr, b1b, nullptr, nullptr, nullptr,
           B, H, H, inpl, pl, 1, 1, 1, 0, 0, 0);
      // t2 = conv3x3(relu(t1)+b2a, stride) + b2b   (fused pre-op gather + async B)
      conv(t1, w2b, t2, nullptr, nullptr, b2a, b2b, nullptr, nullptr, nullptr,
           B, H, H, pl, pl, 3, 3, stride, 1, 1, 0);
      // identity path
      const bf16_t* idn = cur;
      if (wdb) {
        conv(cur, wdb, pool[2], nullptr, nullptr, nullptr, b1b, nullptr, nullptr, nullptr,
             B, H, H, inpl, pl * 4, 1, 1, stride, 0, 0, 0);
        idn = pool[2];
      }
      // x' = relu((conv1x1(relu(t2)+b3a) + b3b)*rscale + rbias + identity)
      conv(t2, w3b, t1, nullptr, idn, b3a, b3b, rscale, rbias, nullptr,
           B, Ho, Ho, pl, pl * 4, 1, 1, 1, 0, 1, 2);

      // rotate ping-pong buffers: new cur = t1 ; old cur becomes scratch
      bf16_t* oldcur = cur;
      cur = t1;
      pool[0] = oldcur;

      H = Ho;
      inpl = pl * 4;
    }
  }

  // ---- head: global average pool (+bias2), then FC as WMMA GEMM
  const float* fcw = (const float*)d_in[ii++];
  const float* fcb = (const float*)d_in[ii++];
  bf16_t* fcwb = wconv(fcw, 1000, 2048, 1, 1);
  {
    const long tot = (long)B * 2048;
    gap_kernel<<<dim3((unsigned)((tot + 255) / 256)), dim3(256), 0, stream>>>(
        cur, gapb, bias2, B, 2048, H * H);
  }
  // FC: M=32 pixels, Kout=1000, Ktot=2048 -> f32 logits straight into d_out
  conv(gapb, fcwb, nullptr, (float*)d_out, nullptr, nullptr, nullptr, nullptr, nullptr, fcb,
       B, 1, 1, 2048, 1000, 1, 1, 1, 0, 0, 4);
}
